// Model_89429809037459
// MI455X (gfx1250) — matrix operand order repaired and checked by reading, not compiled, not run on hardware
//
#include <hip/hip_runtime.h>
#include <cstdint>

// ---------------------------------------------------------------------------
// CDNA5 (gfx1250) HGT forward.  Dense ops use v_wmma_f32_16x16x32_f16
// (fp32 -> f16 inputs, f32 accumulate); A and W tiles staged through LDS so
// fragments are ds_load_b128.  Edge segment-softmax uses order-preserving-int
// atomicMax + native global_atomic_add_f32 (bandwidth bound).
// ---------------------------------------------------------------------------

typedef __attribute__((ext_vector_type(16))) _Float16 v16h;
typedef __attribute__((ext_vector_type(8)))  _Float16 v8h;
typedef __attribute__((ext_vector_type(4)))  _Float16 v4h;
typedef __attribute__((ext_vector_type(8)))  float    v8f;

#define DIMC   128
#define NHEADS 8
#define DHEADC 16
#define LDSTR  40   // padded row stride (halves) for LDS tiles: 80B = 20 banks

__device__ __forceinline__ float gelu_exact(float x) {
  return 0.5f * x * (1.0f + erff(x * 0.7071067811865475f));
}
__device__ __forceinline__ int f32_ord(float f) {
  int i = __float_as_int(f);
  return (i >= 0) ? i : (i ^ 0x7fffffff);
}
__device__ __forceinline__ float ord_f32(int i) {
  int j = (i >= 0) ? i : (i ^ 0x7fffffff);
  return __int_as_float(j);
}
__device__ __forceinline__ float atomAddF(float* p, float v) {
  return unsafeAtomicAdd(p, v);   // native global_atomic_add_f32
}

// Stage A rows [row0,row0+64) x cols [kk,kk+32) into lds_a (f16, optional gelu)
__device__ __forceinline__ void stage_a_tile(
    _Float16 (*lds_a)[LDSTR], const float* __restrict__ A,
    int row0, int M, int K, int kk, int geluA, int tid)
{
  if (row0 + 64 <= M) {               // fast path: unguarded float4 loads
    int r  = tid >> 3;
    int c4 = (tid & 7) * 4;
#pragma unroll
    for (int rr = 0; rr < 64; rr += 32) {
      float4 v = *(const float4*)(A + (size_t)(row0 + r + rr) * K + kk + c4);
      if (geluA) {
        v.x = gelu_exact(v.x); v.y = gelu_exact(v.y);
        v.z = gelu_exact(v.z); v.w = gelu_exact(v.w);
      }
      v4h h;
      h[0] = (_Float16)v.x; h[1] = (_Float16)v.y;
      h[2] = (_Float16)v.z; h[3] = (_Float16)v.w;
      *(v4h*)&lds_a[r + rr][c4] = h;  // 8B-aligned ds_store_b64
    }
  } else {                             // tail block: guarded scalar path
    for (int i = tid; i < 64 * 32; i += 256) {
      int r = i >> 5, c = i & 31;
      int gr = row0 + r;
      float v = 0.f;
      if (gr < M) v = A[(size_t)gr * K + (kk + c)];
      if (geluA) v = gelu_exact(v);
      lds_a[r][c] = (_Float16)v;
    }
  }
}

// Stage W rows [kk,kk+32) x 128 cols TRANSPOSED into lds_wt[col][k] (f16)
__device__ __forceinline__ void stage_w_tile(
    _Float16 (*lds_wt)[LDSTR], const float* __restrict__ W, int kk, int tid)
{
#pragma unroll
  for (int it = 0; it < 4; ++it) {
    int j  = tid + it * 256;          // 0..1023 over (k, c/4)
    int k  = j >> 5;                  // 0..31
    int c4 = (j & 31) * 4;            // 0..124
    float4 w = *(const float4*)(W + (size_t)(kk + k) * DIMC + c4);
    lds_wt[c4 + 0][k] = (_Float16)w.x;
    lds_wt[c4 + 1][k] = (_Float16)w.y;
    lds_wt[c4 + 2][k] = (_Float16)w.z;
    lds_wt[c4 + 3][k] = (_Float16)w.w;
  }
}

// A fragment 16x32 f16 (ISA 7.12.2): elems 0..7 -> K=8*hh+e ; 8..15 -> +16
__device__ __forceinline__ v16h load_a_frag(
    const _Float16 (*lds_a)[LDSTR], int row, int hh)
{
  v8h a0 = *(const v8h*)&lds_a[row][8 * hh];       // K = 8*hh .. 8*hh+7
  v8h a1 = *(const v8h*)&lds_a[row][16 + 8 * hh];  // K = 16+8*hh ..
  v16h af;
#pragma unroll
  for (int e = 0; e < 8; ++e) { af[e] = a0[e]; af[e + 8] = a1[e]; }
  return af;
}

// B fragment 32x16 f16: elems 0..7 -> K=8*hh+e ; 8..15 -> +16
__device__ __forceinline__ v16h load_b_frag(
    const _Float16 (*lds_wt)[LDSTR], int col, int hh)
{
  v8h b0 = *(const v8h*)&lds_wt[col][8 * hh];
  v8h b1 = *(const v8h*)&lds_wt[col][16 + 8 * hh];
  v16h bf;
#pragma unroll
  for (int e = 0; e < 8; ++e) { bf[e] = b0[e]; bf[e + 8] = b1[e]; }
  return bf;
}

// ---------------------------------------------------------------------------
// C[M,128] = act(A[M,K]) @ W[K,128] + bias     (K multiple of 32)
// optional: act = gelu on A;   optional: C = g*C + (1-g)*skip, g=sigmoid(*gate)
// block = 256 threads (8 waves). wave w owns cols [16w,16w+16). 64 rows/block.
// ---------------------------------------------------------------------------
__global__ __launch_bounds__(256) void gemm128_wmma(
    const float* __restrict__ A, int M, int K,
    const float* __restrict__ W, const float* __restrict__ bias,
    float* __restrict__ C, int geluA,
    const float* __restrict__ skip, const float* __restrict__ gate)
{
  __shared__ _Float16 lds_a[64][LDSTR];
  __shared__ _Float16 lds_wt[DIMC][LDSTR];
  const int tid  = threadIdx.x;
  const int wave = tid >> 5;
  const int lane = tid & 31;
  const int hh   = lane >> 4;
  const int lrow = lane & 15;
  const int row0 = blockIdx.x * 64;
  const int col  = wave * 16 + lrow;

  v8f acc[4] = {};

  for (int kk = 0; kk < K; kk += 32) {
    stage_a_tile(lds_a, A, row0, M, K, kk, geluA, tid);
    stage_w_tile(lds_wt, W, kk, tid);
    __syncthreads();

    v16h bf = load_b_frag(lds_wt, col, hh);
#pragma unroll
    for (int rt = 0; rt < 4; ++rt) {
      v16h af = load_a_frag(lds_a, rt * 16 + lrow, hh);
      acc[rt] = __builtin_amdgcn_wmma_f32_16x16x32_f16(
          false, af, false, bf, (short)0, acc[rt], false, false);
    }
    __syncthreads();
  }

  float bb = bias ? bias[col] : 0.f;
  float g = 1.f, gi = 0.f;
  if (skip) { float s = *gate; g = 1.f / (1.f + __expf(-s)); gi = 1.f - g; }
#pragma unroll
  for (int rt = 0; rt < 4; ++rt) {
#pragma unroll
    for (int r = 0; r < 8; ++r) {
      int grow = row0 + rt * 16 + r + 8 * hh;  // C/D layout: M = vgpr + 8*half
      if (grow < M) {
        float v = acc[rt][r] + bb;
        if (skip) v = g * v + gi * skip[(size_t)grow * DIMC + col];
        C[(size_t)grow * DIMC + col] = v;
      }
    }
  }
}

// ---------------------------------------------------------------------------
// Fused K/Q/V: three [M,128]x[128,128] GEMMs sharing one staged A tile.
// 12 accumulators (96 VGPRs), 12 WMMAs per K-step.
// ---------------------------------------------------------------------------
__global__ __launch_bounds__(256) void gemm128_kqv_wmma(
    const float* __restrict__ A, int M,
    const float* __restrict__ W0, const float* __restrict__ W1,
    const float* __restrict__ W2,
    const float* __restrict__ b0, const float* __restrict__ b1,
    const float* __restrict__ b2,
    float* __restrict__ C0, float* __restrict__ C1, float* __restrict__ C2)
{
  __shared__ _Float16 lds_a[64][LDSTR];
  __shared__ _Float16 lds_wt[3][DIMC][LDSTR];
  const int tid  = threadIdx.x;
  const int wave = tid >> 5;
  const int lane = tid & 31;
  const int hh   = lane >> 4;
  const int lrow = lane & 15;
  const int row0 = blockIdx.x * 64;
  const int col  = wave * 16 + lrow;

  v8f acc[3][4] = {};

  for (int kk = 0; kk < DIMC; kk += 32) {
    stage_a_tile(lds_a, A, row0, M, DIMC, kk, 0, tid);
    stage_w_tile(lds_wt[0], W0, kk, tid);
    stage_w_tile(lds_wt[1], W1, kk, tid);
    stage_w_tile(lds_wt[2], W2, kk, tid);
    __syncthreads();

    v16h bf0 = load_b_frag(lds_wt[0], col, hh);
    v16h bf1 = load_b_frag(lds_wt[1], col, hh);
    v16h bf2 = load_b_frag(lds_wt[2], col, hh);
#pragma unroll
    for (int rt = 0; rt < 4; ++rt) {
      v16h af = load_a_frag(lds_a, rt * 16 + lrow, hh);
      acc[0][rt] = __builtin_amdgcn_wmma_f32_16x16x32_f16(
          false, af, false, bf0, (short)0, acc[0][rt], false, false);
      acc[1][rt] = __builtin_amdgcn_wmma_f32_16x16x32_f16(
          false, af, false, bf1, (short)0, acc[1][rt], false, false);
      acc[2][rt] = __builtin_amdgcn_wmma_f32_16x16x32_f16(
          false, af, false, bf2, (short)0, acc[2][rt], false, false);
    }
    __syncthreads();
  }

  const float bb0 = b0[col], bb1 = b1[col], bb2 = b2[col];
#pragma unroll
  for (int rt = 0; rt < 4; ++rt) {
#pragma unroll
    for (int r = 0; r < 8; ++r) {
      int grow = row0 + rt * 16 + r + 8 * hh;
      if (grow < M) {
        size_t o = (size_t)grow * DIMC + col;
        C0[o] = acc[0][rt][r] + bb0;
        C1[o] = acc[1][rt][r] + bb1;
        C2[o] = acc[2][rt][r] + bb2;
      }
    }
  }
}

// ---------------------------------------------------------------------------
// Y[n, h*16+e] = sum_d X[n, h*16+d] * R[h, d, e]   (per-head 16x16, K padded
// to 32 with zeros).  wave = head; 64 nodes per block.
// ---------------------------------------------------------------------------
__global__ __launch_bounds__(256) void rel_transform_wmma(
    const float* __restrict__ X, const float* __restrict__ R,
    float* __restrict__ Y, int N)
{
  const int tid  = threadIdx.x;
  const int h    = tid >> 5;
  const int lane = tid & 31;
  const int hh   = lane >> 4;
  const int lrow = lane & 15;
  const int row0 = blockIdx.x * 64;

  // B = R[h] (16x16), zero-pad K 16..31 (lane-half 1 supplies K>=16 -> zeros)
  v16h bf;
#pragma unroll
  for (int e = 0; e < 16; ++e)
    bf[e] = (hh == 0) ? (_Float16)R[h * 256 + e * 16 + lrow] : (_Float16)0.f;

#pragma unroll
  for (int rt = 0; rt < 4; ++rt) {
    int row = row0 + rt * 16 + lrow;
    v16h af;
#pragma unroll
    for (int e = 0; e < 16; ++e) {   // hh==0 -> K=e in 0..15 ; hh==1 -> pad
      float v = 0.f;
      if (hh == 0 && row < N) v = X[(size_t)row * DIMC + h * DHEADC + e];
      af[e] = (_Float16)v;
    }
    v8f acc = {};
    acc = __builtin_amdgcn_wmma_f32_16x16x32_f16(
        false, af, false, bf, (short)0, acc, false, false);
#pragma unroll
    for (int r = 0; r < 8; ++r) {
      int grow = row0 + rt * 16 + r + 8 * hh;
      if (grow < N) Y[(size_t)grow * DIMC + h * DHEADC + lrow] = acc[r];
    }
  }
}

// ---------------------------------------------------------------------------
// Edge passes (segment softmax over incoming edges per destination)
// ---------------------------------------------------------------------------
__global__ void edge_logits_max_kernel(
    const int* __restrict__ src, const int* __restrict__ dst,
    const float* __restrict__ kr, const float* __restrict__ q,
    const float* __restrict__ prel, float scale,
    float* __restrict__ logits, int* __restrict__ mmax, int E)
{
  int e = blockIdx.x * blockDim.x + threadIdx.x;
  if (e >= E) return;
  int s = src[e], d = dst[e];
  const float4* kp = (const float4*)(kr + (size_t)s * DIMC);
  const float4* qp = (const float4*)(q  + (size_t)d * DIMC);
#pragma unroll
  for (int h = 0; h < NHEADS; ++h) {
    float acc = 0.f;
#pragma unroll
    for (int j = 0; j < 4; ++j) {
      float4 a = kp[h * 4 + j], b = qp[h * 4 + j];
      acc += a.x * b.x + a.y * b.y + a.z * b.z + a.w * b.w;
    }
    float lg = acc * prel[h] * scale;
    logits[(size_t)e * NHEADS + h] = lg;
    atomicMax(&mmax[d * NHEADS + h], f32_ord(lg));
  }
}

__global__ void edge_exp_sum_kernel(
    const int* __restrict__ dst, float* __restrict__ logits,
    const int* __restrict__ mmax, float* __restrict__ ssum, int E)
{
  int e = blockIdx.x * blockDim.x + threadIdx.x;
  if (e >= E) return;
  int d = dst[e];
#pragma unroll
  for (int h = 0; h < NHEADS; ++h) {
    float m  = ord_f32(mmax[d * NHEADS + h]);
    float ev = __expf(logits[(size_t)e * NHEADS + h] - m);
    logits[(size_t)e * NHEADS + h] = ev;
    atomAddF(&ssum[d * NHEADS + h], ev);
  }
}

__global__ void edge_aggregate_kernel(
    const int* __restrict__ src, const int* __restrict__ dst,
    const float* __restrict__ logits, const float* __restrict__ ssum,
    const float* __restrict__ vr, float* __restrict__ agg, int E)
{
  int e = blockIdx.x * blockDim.x + threadIdx.x;
  if (e >= E) return;
  int s = src[e], d = dst[e];
#pragma unroll
  for (int h = 0; h < NHEADS; ++h) {
    float alpha = logits[(size_t)e * NHEADS + h] / (ssum[d * NHEADS + h] + 1e-16f);
    const float4* vp = (const float4*)(vr + (size_t)s * DIMC + h * DHEADC);
    float* ap = agg + (size_t)d * DIMC + h * DHEADC;
#pragma unroll
    for (int j = 0; j < 4; ++j) {
      float4 v = vp[j];
      atomAddF(&ap[j * 4 + 0], v.x * alpha);
      atomAddF(&ap[j * 4 + 1], v.y * alpha);
      atomAddF(&ap[j * 4 + 2], v.z * alpha);
      atomAddF(&ap[j * 4 + 3], v.w * alpha);
    }
  }
}

// ---------------------------------------------------------------------------
__global__ void classifier_kernel(
    const int* __restrict__ mi, const int* __restrict__ di,
    const float* __restrict__ xm, const float* __restrict__ xd,
    const float* __restrict__ Wc, const float* __restrict__ bc,
    float* __restrict__ out, int E)
{
  int e = blockIdx.x * blockDim.x + threadIdx.x;
  if (e >= E) return;
  float acc = bc[0];
  const float4* a  = (const float4*)(xm + (size_t)mi[e] * DIMC);
  const float4* b  = (const float4*)(xd + (size_t)di[e] * DIMC);
  const float4* w0 = (const float4*)Wc;
  const float4* w1 = (const float4*)(Wc + DIMC);
#pragma unroll 8
  for (int j = 0; j < 32; ++j) {
    float4 av = a[j], wv = w0[j];
    acc += av.x * wv.x + av.y * wv.y + av.z * wv.z + av.w * wv.w;
  }
#pragma unroll 8
  for (int j = 0; j < 32; ++j) {
    float4 bv = b[j], wv = w1[j];
    acc += bv.x * wv.x + bv.y * wv.y + bv.z * wv.z + bv.w * wv.w;
  }
  out[e] = 1.f / (1.f + __expf(-acc));
}

__global__ void fill_f32(float* __restrict__ p, float v, int n) {
  int i = blockIdx.x * blockDim.x + threadIdx.x;
  if (i < n) p[i] = v;
}
__global__ void fill_i32(int* __restrict__ p, int v, int n) {
  int i = blockIdx.x * blockDim.x + threadIdx.x;
  if (i < n) p[i] = v;
}

// ---------------------------------------------------------------------------
// Host orchestration.
// d_in[0..12] = data inputs in setup_inputs() insertion order; d_in[13..]
// = params leaves in jax pytree (recursively sorted-key) order.
// ---------------------------------------------------------------------------
extern "C" void kernel_launch(void* const* d_in, const int* in_sizes, int n_in,
                              void* d_out, int out_size, void* d_ws, size_t ws_size,
                              hipStream_t stream)
{
  (void)n_in; (void)ws_size;
  // node types: 0=mirna, 1=disease, 2=pcg
  const int IN_DIM[3] = {640, 768, 768};
  const float* x_in[3] = {(const float*)d_in[0], (const float*)d_in[1], (const float*)d_in[2]};
  int N[3];
  for (int t = 0; t < 3; ++t) N[t] = in_sizes[t] / IN_DIM[t];

  // sorted edge-key order -> {edge-index input slot, src type, dst type}
  struct EdgeMeta { int ei, st, dt; };
  const EdgeMeta EM[9] = {
    {4, 1, 1},   // disease__fatherson__disease
    {11, 1, 0},  // disease__rev_association__mirna
    {9, 1, 2},   // disease__rev_association__pcg
    {10, 0, 1},  // mirna__association__disease
    {6, 0, 2},   // mirna__association__pcg
    {3, 0, 0},   // mirna__family__mirna
    {8, 2, 1},   // pcg__association__disease
    {5, 2, 2},   // pcg__interaction__pcg
    {7, 2, 0},   // pcg__rev_association__mirna
  };
  int Ecnt[9], maxE = 0;
  for (int i = 0; i < 9; ++i) { Ecnt[i] = in_sizes[EM[i].ei] / 2; if (Ecnt[i] > maxE) maxE = Ecnt[i]; }
  const int Ecls = in_sizes[12] / 2;

  // ---- params leaf indices (tree order: cls < emb < layers; sorted keys) ----
  int p = 13;
  const int clsW = p++; const int clsB = p++;
  int embW[3], embB[3];
  const int nodeOrder[3] = {1, 0, 2};            // disease < mirna < pcg
  for (int j = 0; j < 3; ++j) { int t = nodeOrder[j]; embW[t] = p++; embB[t] = p++; }
  struct NodeP { int Wa, Wk, Wq, Wv, ba, bk, bq, bv, skip; };
  struct EdgeP { int a_rel, m_rel, p_rel; };
  EdgeP EP[2][9]; NodeP NP[2][3];
  for (int l = 0; l < 2; ++l) {
    for (int e = 0; e < 9; ++e) { EP[l][e].a_rel = p++; EP[l][e].m_rel = p++; EP[l][e].p_rel = p++; }
    for (int j = 0; j < 3; ++j) {
      int t = nodeOrder[j]; NodeP& q = NP[l][t];
      q.Wa = p++; q.Wk = p++; q.Wq = p++; q.Wv = p++;
      q.ba = p++; q.bk = p++; q.bq = p++; q.bv = p++; q.skip = p++;
    }
  }

  // ---- workspace carve ----
  char* wp = (char*)d_ws;
  auto carve = [&](size_t bytes) -> void* {
    void* r = (void*)wp; wp += (bytes + 255) & ~(size_t)255; return r;
  };
  float *X[3], *XN[3], *Kb[3], *Qb[3], *Vb[3], *Ag[3];
  for (int t = 0; t < 3; ++t) X[t]  = (float*)carve((size_t)N[t] * DIMC * 4);
  for (int t = 0; t < 3; ++t) XN[t] = (float*)carve((size_t)N[t] * DIMC * 4);
  for (int t = 0; t < 3; ++t) Kb[t] = (float*)carve((size_t)N[t] * DIMC * 4);
  for (int t = 0; t < 3; ++t) Qb[t] = (float*)carve((size_t)N[t] * DIMC * 4);
  for (int t = 0; t < 3; ++t) Vb[t] = (float*)carve((size_t)N[t] * DIMC * 4);
  for (int t = 0; t < 3; ++t) Ag[t] = (float*)carve((size_t)N[t] * DIMC * 4);
  int maxN = N[0]; if (N[1] > maxN) maxN = N[1]; if (N[2] > maxN) maxN = N[2];
  float* KR  = (float*)carve((size_t)maxN * DIMC * 4);
  float* VR  = (float*)carve((size_t)maxN * DIMC * 4);
  int*   Mb  = (int*)  carve((size_t)maxN * NHEADS * 4);
  float* Sb  = (float*)carve((size_t)maxN * NHEADS * 4);
  float* EL  = (float*)carve((size_t)maxE * NHEADS * 4);

  auto F = [&](int i) { return (const float*)d_in[i]; };
  auto gemm = [&](const float* A, int M, int K, const float* W, const float* bias,
                  float* C, int geluA, const float* skip, const float* gate) {
    gemm128_wmma<<<dim3((M + 63) / 64), 256, 0, stream>>>(A, M, K, W, bias, C, geluA, skip, gate);
  };
  const float scale = 0.25f;   // 1/sqrt(DHEAD=16)

  // ---- input embedding ----
  for (int t = 0; t < 3; ++t)
    gemm(x_in[t], N[t], IN_DIM[t], F(embW[t]), F(embB[t]), X[t], 0, nullptr, nullptr);

  float* cur[3] = {X[0], X[1], X[2]};
  float* nxt[3] = {XN[0], XN[1], XN[2]};

  // ---- layers ----
  for (int l = 0; l < 2; ++l) {
    for (int t = 0; t < 3; ++t) {
      const NodeP& q = NP[l][t];
      gemm128_kqv_wmma<<<dim3((N[t] + 63) / 64), 256, 0, stream>>>(
          cur[t], N[t],
          F(q.Wk), F(q.Wq), F(q.Wv), F(q.bk), F(q.bq), F(q.bv),
          Kb[t], Qb[t], Vb[t]);
      int n = N[t] * DIMC;
      fill_f32<<<dim3((n + 255) / 256), 256, 0, stream>>>(Ag[t], 0.f, n);
    }
    for (int e = 0; e < 9; ++e) {
      const int st = EM[e].st, dt = EM[e].dt, E = Ecnt[e];
      const int* src = (const int*)d_in[EM[e].ei];
      const int* dst = src + E;
      rel_transform_wmma<<<dim3((N[st] + 63) / 64), 256, 0, stream>>>(Kb[st], F(EP[l][e].a_rel), KR, N[st]);
      rel_transform_wmma<<<dim3((N[st] + 63) / 64), 256, 0, stream>>>(Vb[st], F(EP[l][e].m_rel), VR, N[st]);
      int nd = N[dt] * NHEADS;
      fill_i32<<<dim3((nd + 255) / 256), 256, 0, stream>>>(Mb, (int)0x80000000, nd);
      fill_f32<<<dim3((nd + 255) / 256), 256, 0, stream>>>(Sb, 0.f, nd);
      dim3 eg((E + 255) / 256);
      edge_logits_max_kernel<<<eg, 256, 0, stream>>>(src, dst, KR, Qb[dt], F(EP[l][e].p_rel), scale, EL, Mb, E);
      edge_exp_sum_kernel  <<<eg, 256, 0, stream>>>(dst, EL, Mb, Sb, E);
      edge_aggregate_kernel<<<eg, 256, 0, stream>>>(src, dst, EL, Sb, VR, Ag[dt], E);
    }
    for (int t = 0; t < 3; ++t) {
      const NodeP& q = NP[l][t];
      gemm(Ag[t], N[t], DIMC, F(q.Wa), F(q.ba), nxt[t], /*gelu=*/1, /*skip=*/cur[t], /*gate=*/F(q.skip));
    }
    for (int t = 0; t < 3; ++t) { float* tmp = cur[t]; cur[t] = nxt[t]; nxt[t] = tmp; }
  }

  // ---- classifier ----
  const int* mi = (const int*)d_in[12];
  const int* di = mi + Ecls;
  classifier_kernel<<<dim3((Ecls + 255) / 256), 256, 0, stream>>>(
      mi, di, cur[0], cur[1], F(clsW), F(clsB), (float*)d_out, Ecls);
  (void)out_size;
}
